// S4D_28853590294580
// MI455X (gfx1250) — matrix operand order repaired and checked by reading, not compiled, not run on hardware
//
#include <hip/hip_runtime.h>
#include <hip/hip_bf16.h>
#include <cmath>

// ---------------------------------------------------------------------------
// S4D forward for MI455X (gfx1250, wave32).
//  - FFT conv replaced by chunked linear-recurrence scan (u-bandwidth bound).
//  - Output 1x1 conv = bf16 WMMA GEMM, LDS-staged with BOTH gfx1250 copy
//    engines: per-thread global_load_async_to_lds_b128 (ASYNCcnt) for the A
//    tile and a wave-issued TDM tensor_load_to_lds (TENSORcnt) for the B
//    tile, double-buffered so copies overlap WMMA.
// ---------------------------------------------------------------------------

#define H_DIM 1024
#define N2    32
#define B_DIM 8
#define L_DIM 2048
#define NCOL  (B_DIM * L_DIM)   // 16384 GEMM columns
#define LDS_PITCH 40            // ushorts per row (80B) -> conflict-free frags

typedef __attribute__((ext_vector_type(16))) __bf16 v16bf;
typedef __attribute__((ext_vector_type(8)))  __bf16 v8bf;
typedef __attribute__((ext_vector_type(8)))  float  v8f;
typedef __attribute__((ext_vector_type(4)))  unsigned int v4u;
typedef __attribute__((ext_vector_type(8)))  unsigned int v8u;

__device__ __forceinline__ unsigned short f32_to_bf16(float f) {
    unsigned int x = __float_as_uint(f);
    unsigned int r = x + 0x7FFFu + ((x >> 16) & 1u);   // round-to-nearest-even
    return (unsigned short)(r >> 16);
}

// Low 32 bits of a generic LDS pointer = workgroup-relative LDS byte offset.
__device__ __forceinline__ unsigned lds_off(const void* p) {
    return (unsigned)(size_t)p;
}

// 16-byte async copy global -> LDS (ASYNCcnt).
__device__ __forceinline__ void async_copy_b128(unsigned lds_dst, const void* gsrc) {
    asm volatile("global_load_async_to_lds_b128 %0, %1, off"
                 :: "v"(lds_dst), "v"(gsrc) : "memory");
}
__device__ __forceinline__ void wait_async0() {
    asm volatile("s_wait_asynccnt 0x0" ::: "memory");
}

// TDM: load a 2D tile 32(K) x 128(rows) of 2B elements, row stride H_DIM,
// into LDS with hardware padding 16B per 64B row (=> LDS_PITCH 80B rows).
// D# built per CDNA5 ISA; groups 2/3 zero.
__device__ __forceinline__ void tdm_load_b_tile(const unsigned short* gsrc,
                                                unsigned ldsDst) {
    const unsigned long long ga = (unsigned long long)(size_t)gsrc;
    v4u g0;
    g0[0] = 1u;                                             // count=1
    g0[1] = ldsDst;                                         // lds_addr (bytes)
    g0[2] = (unsigned)(ga & 0xFFFFFFFFull);                 // global_addr lo
    g0[3] = (unsigned)((ga >> 32) & 0x1FFFFFFull) | (2u << 30);  // hi | type=2
    v8u g1;
    g1[0] = (1u << 16)        // data_size = 2B
          | (1u << 20)        // pad_enable
          | (3u << 22)        // pad_interval: every 16 DWORDs (64B)
          | (3u << 25);       // pad_amount: 4 DWORDs (16B)
    g1[1] = (unsigned)H_DIM << 16;            // tensor_dim0 = 1024 (lo16@63:48)
    g1[2] = ((unsigned)NCOL & 0xFFFFu) << 16; // dim0 hi=0 | tensor_dim1 lo16
    g1[3] = ((unsigned)NCOL >> 16) | (32u << 16);  // dim1 hi | tile_dim0=32
    g1[4] = 128u;                             // tile_dim1=128, tile_dim2=0
    g1[5] = (unsigned)H_DIM;                  // tensor_dim0_stride lo32
    g1[6] = 0u;                               // stride hi | dim1_stride lo
    g1[7] = 0u;
    const v4u gz = {0u, 0u, 0u, 0u};
    asm volatile("tensor_load_to_lds %0, %1, %2, %3"
                 :: "s"(g0), "s"(g1), "s"(gz), "s"(gz) : "memory");
}

// ---------------------------------------------------------------------------
// Kernel 1: per-(h,n) pole A = exp(dt*w), coefficient C' = C*(A-1)/w, and the
// chunked-scan tables:
//   ktab[h][m]    = 2*Re(sum_n C'_n A_n^m)            (intra-chunk taps)
//   Wtab[h][j][n] = 2*C'_n*A_n^(j+1)  (complex)       (state -> output)
// ---------------------------------------------------------------------------
__global__ __launch_bounds__(32) void s4d_setup(
    const float* __restrict__ log_dt, const float* __restrict__ log_w_real,
    const float* __restrict__ w_imag, const float* __restrict__ C_re,
    const float* __restrict__ C_im, float* __restrict__ AC,
    float* __restrict__ ktab, float2* __restrict__ Wtab)
{
    const int h = blockIdx.x;
    const int n = threadIdx.x;                 // 32 modes, one per lane
    const float dt = __expf(log_dt[h]);
    const float wr = -__expf(log_w_real[n]);   // w = -exp(log_w_real) + i*w_imag
    const float wi = w_imag[n];
    const float er = __expf(wr * dt);
    const float th = wi * dt;
    const float Ar = er * __cosf(th);
    const float Ai = er * __sinf(th);
    const float Er = Ar - 1.0f, Ei = Ai;       // (A - 1) / w via conjugate
    const float inv = 1.0f / (wr * wr + wi * wi);
    const float qr = (Er * wr + Ei * wi) * inv;
    const float qi = (Ei * wr - Er * wi) * inv;
    const float cr = C_re[h * N2 + n], ci = C_im[h * N2 + n];
    const float Cr = cr * qr - ci * qi;        // C'_re
    const float Ci = cr * qi + ci * qr;        // C'_im
    float4 v; v.x = Ar; v.y = Ai; v.z = Cr; v.w = Ci;
    *(float4*)(AC + ((size_t)h * N2 + n) * 4) = v;

    // ktab: iterate t = C'*A^m, reduce Re over modes.
    float tr = Cr, ti = Ci;
    for (int m = 0; m < 32; ++m) {
        float red = tr;
        red += __shfl_xor(red, 16, 32);
        red += __shfl_xor(red,  8, 32);
        red += __shfl_xor(red,  4, 32);
        red += __shfl_xor(red,  2, 32);
        red += __shfl_xor(red,  1, 32);
        if (n == 0) ktab[h * 32 + m] = 2.0f * red;
        const float t2r = tr * Ar - ti * Ai;
        const float t2i = tr * Ai + ti * Ar;
        tr = t2r; ti = t2i;
    }
    // Wtab: p = C'*A^(j+1)
    float pr = Cr * Ar - Ci * Ai;
    float pi = Cr * Ai + Ci * Ar;
    for (int j = 0; j < 32; ++j) {
        Wtab[((size_t)h * 32 + j) * 32 + n] = make_float2(2.0f * pr, 2.0f * pi);
        const float t2r = pr * Ar - pi * Ai;
        const float t2i = pr * Ai + pi * Ar;
        pr = t2r; pi = t2i;
    }
}

// ---------------------------------------------------------------------------
// Kernel 2: convert W_out to bf16 (2MB; lives in the 192MB L2 afterwards).
// ---------------------------------------------------------------------------
__global__ void cvt_bf16(const float* __restrict__ src,
                         unsigned short* __restrict__ dst, int n)
{
    int i = blockIdx.x * blockDim.x + threadIdx.x;
    if (i < n) dst[i] = f32_to_bf16(src[i]);
}

// ---------------------------------------------------------------------------
// Kernel 3: chunked SSM scan.  One wave32 per (b,h); chunk Q = 32.
//  Phase 1 (lane = mode n):  Horner-advance state through the chunk's 32
//                            inputs — no cross-lane traffic.
//  Phase 2 (lane = pos j):   y[j] = sum_n Wtab[h][j][n] (.) s_prev[n]
//                                 + sum_{m<=j} ktab[h][m]*u[j-m] + D_h*u[j]
//  Then exact GELU, bf16, transposed store Xt[(b*L+l)*H + h].
// ---------------------------------------------------------------------------
__global__ __launch_bounds__(32) void s4d_scan(
    const float* __restrict__ u, const float* __restrict__ AC,
    const float* __restrict__ ktab, const float2* __restrict__ Wtab,
    const float* __restrict__ D, unsigned short* __restrict__ Xt)
{
    const int bh = blockIdx.x;
    const int b  = bh >> 10;           // / H_DIM
    const int h  = bh & (H_DIM - 1);
    const int lane = threadIdx.x;

    __shared__ float  su[32];          // current chunk of u
    __shared__ float2 ss[32];          // s_prev (state at chunk entry)
    __shared__ float  sk[32];          // intra-chunk taps

    const float4 ac = *(const float4*)(AC + ((size_t)h * N2 + lane) * 4);
    const float Ar = ac.x, Ai = ac.y;
    sk[lane] = ktab[h * 32 + lane];

    // This lane's output row of Wtab (j = lane), kept in registers.
    float wre[32], wim[32];
    const float2* wrow = Wtab + ((size_t)h * 32 + lane) * 32;
#pragma unroll
    for (int n2 = 0; n2 < 32; ++n2) { const float2 w = wrow[n2]; wre[n2] = w.x; wim[n2] = w.y; }

    const float Dh = D[h];
    const float* __restrict__ urow = u + (size_t)bh * L_DIM;
    unsigned short* __restrict__ xcol = Xt + (size_t)b * L_DIM * H_DIM + h;

    float sr = 0.0f, si = 0.0f;        // state of mode `lane`
    for (int c = 0; c < L_DIM / 32; ++c) {
        __syncthreads();
        su[lane] = __builtin_nontemporal_load(urow + c * 32 + lane);
        ss[lane] = make_float2(sr, si);
        __syncthreads();
        // Phase 1: lane = mode. 32 sequential complex FMAs, no shuffles.
#pragma unroll
        for (int i = 0; i < 32; ++i) {
            const float uv = su[i];
            const float t = fmaf(Ar, sr, fmaf(-Ai, si, uv));
            si = fmaf(Ar, si, Ai * sr);
            sr = t;
        }
        // Phase 2: lane = position j.
        float acc = 0.0f;
#pragma unroll
        for (int n2 = 0; n2 < 32; ++n2) {
            const float2 sv = ss[n2];                    // broadcast read
            acc = fmaf(wre[n2],  sv.x, acc);
            acc = fmaf(-wim[n2], sv.y, acc);
        }
#pragma unroll
        for (int m = 0; m < 32; ++m) {
            const float kf = (m <= lane) ? sk[m] : 0.0f; // causal mask
            acc = fmaf(kf, su[(lane - m) & 31], acc);    // conflict-free
        }
        const float y = fmaf(Dh, su[lane], acc);
        const float g = 0.5f * y * (1.0f + erff(y * 0.70710678118654752f));
        xcol[(size_t)(c * 32 + lane) * H_DIM] = f32_to_bf16(g);
    }
}

// ---------------------------------------------------------------------------
// Kernel 4: out = W_out(bf16) @ Xt(bf16) + b_out via v_wmma_f32_16x16x32_bf16.
// M=1024 (o), K=1024 (i), N=16384 (b*L).  Block = 8 waves (2M x 4N): tile
// 64x128, K-tile 32, double-buffered LDS:
//   A tile (64x32) : 1 global_load_async_to_lds_b128 per thread  (ASYNCcnt)
//   B tile (128x32): 1 tensor_load_to_lds issued by wave 0        (TENSORcnt)
// Fragment layouts per CDNA5 ISA 7.12.2 (half = lane>>4, cl = lane&15).
// ---------------------------------------------------------------------------
__global__ __launch_bounds__(256) void out_gemm(
    const unsigned short* __restrict__ Wb,   // [H][H] bf16 (row = o, col = i)
    const unsigned short* __restrict__ Xt,   // [NCOL][H] bf16 (row = col, col = i)
    const float* __restrict__ b_out,
    float* __restrict__ out)                  // [B][H][L] f32
{
    __shared__ unsigned short lA[2][64  * LDS_PITCH];
    __shared__ unsigned short lB[2][128 * LDS_PITCH];

    const int tid  = threadIdx.x;
    const int wave = tid >> 5;
    const int lane = tid & 31;
    const int cl   = lane & 15;
    const int half = lane >> 4;

    const int oBlock   = blockIdx.y * 64;
    const int colBlock = blockIdx.x * 128;
    const int oBase    = oBlock   + (wave >> 2) * 32;
    const int colBase  = colBlock + (wave & 3)  * 32;

    // A-tile async-copy assignment: 64 rows x 4 x 16B chunks = 1 per thread.
    const int aRow = tid >> 2, aChunk = tid & 3;
    const unsigned aOffB = (unsigned)(aRow * LDS_PITCH + aChunk * 8) * 2u;
    const unsigned aDst[2] = { lds_off(&lA[0][0]) + aOffB,
                               lds_off(&lA[1][0]) + aOffB };
    const unsigned short* aRowG = Wb + (size_t)(oBlock + aRow) * H_DIM + aChunk * 8;
    const unsigned short* bTileG = Xt + (size_t)colBlock * H_DIM;

    v8f acc[2][2];
#pragma unroll
    for (int tm = 0; tm < 2; ++tm) {
#pragma unroll
        for (int j = 0; j < 8; ++j) {
            const float bo = b_out[oBase + tm * 16 + j + 8 * half];
            acc[tm][0][j] = bo;
            acc[tm][1][j] = bo;
        }
    }

    // Prologue: fill buffer 0.
    async_copy_b128(aDst[0], aRowG);
    if (wave == 0) tdm_load_b_tile(bTileG, lds_off(&lB[0][0]));

    int pb = 0;
    for (int kb = 0; kb < H_DIM; kb += 32, pb ^= 1) {
        wait_async0();                                   // my A copies landed
        if (wave == 0) __builtin_amdgcn_s_wait_tensorcnt(0);  // B tile landed
        __syncthreads();                                 // everyone's landed
        if (kb + 32 < H_DIM) {                           // prefetch next tile
            async_copy_b128(aDst[pb ^ 1], aRowG + kb + 32);
            if (wave == 0)
                tdm_load_b_tile(bTileG + kb + 32, lds_off(&lB[pb ^ 1][0]));
        }

        v16bf afr[2];
#pragma unroll
        for (int tm = 0; tm < 2; ++tm) {
            const unsigned short* ar =
                &lA[pb][((wave >> 2) * 32 + tm * 16 + cl) * LDS_PITCH];
            const v8bf a0 = *(const v8bf*)(ar + half * 8);
            const v8bf a1 = *(const v8bf*)(ar + 16 + half * 8);
#pragma unroll
            for (int e = 0; e < 8; ++e) { afr[tm][e] = a0[e]; afr[tm][e + 8] = a1[e]; }
        }
        v16bf bfr[2];
#pragma unroll
        for (int tn = 0; tn < 2; ++tn) {
            const unsigned short* br =
                &lB[pb][((wave & 3) * 32 + tn * 16 + cl) * LDS_PITCH];
            const v8bf b0 = *(const v8bf*)(br + half * 8);
            const v8bf b1 = *(const v8bf*)(br + 16 + half * 8);
#pragma unroll
            for (int e = 0; e < 8; ++e) { bfr[tn][e] = b0[e]; bfr[tn][e + 8] = b1[e]; }
        }
#pragma unroll
        for (int tm = 0; tm < 2; ++tm)
#pragma unroll
            for (int tn = 0; tn < 2; ++tn)
                acc[tm][tn] = __builtin_amdgcn_wmma_f32_16x16x32_bf16(
                    false, afr[tm], false, bfr[tn],
                    (short)0, acc[tm][tn], false, false);
    }

    // Scatter stores per C/D layout: out[b][o][l], col = b*L + l (NT writes).
#pragma unroll
    for (int tm = 0; tm < 2; ++tm) {
#pragma unroll
        for (int tn = 0; tn < 2; ++tn) {
            const int col  = colBase + tn * 16 + cl;
            const int bidx = col >> 11;            // / L_DIM
            const int lpos = col & (L_DIM - 1);
#pragma unroll
            for (int j = 0; j < 8; ++j) {
                const int o = oBase + tm * 16 + j + 8 * half;
                __builtin_nontemporal_store(
                    acc[tm][tn][j],
                    &out[((size_t)bidx * H_DIM + o) * L_DIM + lpos]);
            }
        }
    }
}

// ---------------------------------------------------------------------------
// Launcher.  Workspace layout (needs ~43 MB):
//   [0,      512KB)  AC table   (H*N2*4 f32)
//   [512KB,  640KB)  ktab       (H*32 f32)
//   [1MB,    9MB)    Wtab       (H*32*32 float2)
//   [9MB,    11MB)   W_out bf16 (H*H)
//   [11MB,   43MB)   Xt activations bf16 (NCOL*H)
// ---------------------------------------------------------------------------
extern "C" void kernel_launch(void* const* d_in, const int* in_sizes, int n_in,
                              void* d_out, int out_size, void* d_ws, size_t ws_size,
                              hipStream_t stream)
{
    const float* u          = (const float*)d_in[0];
    const float* log_dt     = (const float*)d_in[1];
    const float* log_w_real = (const float*)d_in[2];
    const float* w_imag     = (const float*)d_in[3];
    const float* C_re       = (const float*)d_in[4];
    const float* C_im       = (const float*)d_in[5];
    const float* D          = (const float*)d_in[6];
    const float* W_out      = (const float*)d_in[7];
    const float* b_out      = (const float*)d_in[8];
    float* out              = (float*)d_out;

    char* ws = (char*)d_ws;
    float*          AC   = (float*)ws;
    float*          ktab = (float*)(ws + (size_t)512 * 1024);
    float2*         Wtab = (float2*)(ws + (size_t)1024 * 1024);
    unsigned short* Wb   = (unsigned short*)(ws + (size_t)9 * 1024 * 1024);
    unsigned short* Xt   = (unsigned short*)(ws + (size_t)11 * 1024 * 1024);

    s4d_setup<<<H_DIM, 32, 0, stream>>>(log_dt, log_w_real, w_imag, C_re, C_im,
                                        AC, ktab, Wtab);
    cvt_bf16<<<(H_DIM * H_DIM + 255) / 256, 256, 0, stream>>>(W_out, Wb, H_DIM * H_DIM);
    s4d_scan<<<B_DIM * H_DIM, 32, 0, stream>>>(u, AC, ktab, Wtab, D, Xt);
    dim3 g(NCOL / 128, H_DIM / 64);
    out_gemm<<<g, 256, 0, stream>>>(Wb, Xt, b_out, out);
}
